// TransformerDecoderLayerAsGNN_9663676416889
// MI455X (gfx1250) — matrix operand order repaired and checked by reading, not compiled, not run on hardware
//
#include <hip/hip_runtime.h>

// ---------------- problem constants (match reference) ----------------
#define B_   4
#define T_   1024
#define S_   1024
#define E_   512
#define H_   8
#define F_   2048
#define D_   64
#define EPS_ 1e-5f
#define SCALE_ 0.125f   // D^-0.5 = 64^-0.5

typedef __attribute__((ext_vector_type(16))) __bf16 v16bf;
typedef __attribute__((ext_vector_type(8)))  float  v8f;

static __device__ __forceinline__ v8f wmma_bf16(v16bf a, v16bf b, v8f c) {
  // D = A(16x32 bf16) * B(32x16 bf16) + C(16x16 f32)
  return __builtin_amdgcn_wmma_f32_16x16x32_bf16(false, a, false, b, (short)0, c,
                                                 false, false);
}

// pack two f32 -> one u32 holding {bf16(a) | bf16(b)<<16}  (v_cvt_pk_bf16_f32)
static __device__ __forceinline__ unsigned int pack_bf16(float a, float b) {
  union { __bf16 h[2]; unsigned int u; } p;
  p.h[0] = (__bf16)a; p.h[1] = (__bf16)b;
  return p.u;
}

// 8 consecutive LDS words (16B-aligned) -> v16bf fragment
static __device__ __forceinline__ v16bf load_frag8(const unsigned int* p) {
  union { uint4 q[2]; v16bf v; } u;
  u.q[0] = *(const uint4*)p;
  u.q[1] = *(const uint4*)(p + 4);
  return u.v;
}
// two separated 16B chunks -> v16bf fragment (A-matrix layout)
static __device__ __forceinline__ v16bf load_frag4x2(const unsigned int* p0,
                                                     const unsigned int* p1) {
  union { uint4 q[2]; v16bf v; } u;
  u.q[0] = *(const uint4*)p0;
  u.q[1] = *(const uint4*)p1;
  return u.v;
}

// =====================================================================
// GEMM: C[M,N] = A[M,K] @ B[K,N] + bias (+ residual) (+ relu)
// f32 global, bf16 WMMA, f32 accumulate. Block tile 128x64, BK=32,
// 256 threads = 8 waves (4M x 2N); wave = 32x32 = 2x2 WMMA frags.
// LDS tiles are bf16-pair-packed u32 word grids:
//   Asw[row][word]  word w  = elements k=2w,2w+1   (16 words/row, stride 20)
//   Bsw[col][word]  word w  = elements k=2w,2w+1   (k-pair-major transpose)
// Strides are multiples of 4 words => every fragment chunk is 16B-aligned
// => fragment loads become ds_load_b128 pairs, no sub-word repacking.
// =====================================================================
#define BM 128
#define BN 64
#define BK 32
#define ALDW 20   // words per A row   (16 + 4 pad), 80B: 16B-aligned
#define BLDW 20   // words per B column(16 + 4 pad)

__global__ __launch_bounds__(256)
void gemm_bf16_wmma(const float* __restrict__ A, const float* __restrict__ Bw,
                    const float* __restrict__ bias, const float* __restrict__ res,
                    float* __restrict__ C, int M, int N, int K, int relu)
{
  __shared__ unsigned int Asw[BM * ALDW];   // 10.0 KB
  __shared__ unsigned int Bsw[BN * BLDW];   //  5.0 KB

  const int tid  = threadIdx.x;
  const int wave = tid >> 5, lane = tid & 31;
  const int wm   = wave >> 1, wn = wave & 1;
  const int mrow = lane & 15, half = lane >> 4;
  const int m0 = blockIdx.y * BM, n0 = blockIdx.x * BN;

  v8f acc[2][2];
#pragma unroll
  for (int i = 0; i < 2; ++i)
#pragma unroll
    for (int j = 0; j < 2; ++j)
#pragma unroll
      for (int r = 0; r < 8; ++r) acc[i][j][r] = 0.f;

  for (int k0 = 0; k0 < K; k0 += BK) {
    // ---- stage A tile (128x32) -> pair-packed words ----
    {
      const int c = (tid & 7) * 4;              // float col within tile
#pragma unroll
      for (int p = 0; p < 4; ++p) {
        const int r = p * 32 + (tid >> 3);
        const float4 f = *(const float4*)(A + (size_t)(m0 + r) * K + (k0 + c));
        uint2 w; w.x = pack_bf16(f.x, f.y); w.y = pack_bf16(f.z, f.w);
        *(uint2*)&Asw[r * ALDW + (c >> 1)] = w;
      }
      if (k0 + BK < K)
        __builtin_prefetch(A + (size_t)(m0 + (tid >> 3)) * K + (k0 + BK + c), 0, 0);
    }
    // ---- stage B tile (32x64) -> k-pair-major transposed words ----
    {
      const int n  = tid & 63;
      const int wb = tid >> 6;                  // 0..3
#pragma unroll
      for (int p = 0; p < 4; ++p) {
        const int w = p * 4 + wb;               // k-word 0..15
        const float f0 = Bw[(size_t)(k0 + 2 * w)     * N + (n0 + n)];
        const float f1 = Bw[(size_t)(k0 + 2 * w + 1) * N + (n0 + n)];
        Bsw[n * BLDW + w] = pack_bf16(f0, f1);
      }
      if (k0 + BK < K)
        __builtin_prefetch(Bw + (size_t)(k0 + BK + 2 * wb) * N + (n0 + n), 0, 0);
    }
    __syncthreads();

    // ---- fragments: contiguous word runs -> ds_load_b128 ----
    v16bf af[2], bfg[2];
#pragma unroll
    for (int i = 0; i < 2; ++i) {
      const unsigned int* ab = &Asw[(wm * 32 + i * 16 + mrow) * ALDW];
      af[i] = load_frag4x2(ab + (half << 2), ab + 8 + (half << 2));
    }
#pragma unroll
    for (int j = 0; j < 2; ++j) {
      const int nn = wn * 32 + j * 16 + mrow;
      bfg[j] = load_frag4x2(&Bsw[nn * BLDW + (half << 2)], &Bsw[nn * BLDW + 8 + (half << 2)]);
    }
#pragma unroll
    for (int i = 0; i < 2; ++i)
#pragma unroll
      for (int j = 0; j < 2; ++j)
        acc[i][j] = wmma_bf16(af[i], bfg[j], acc[i][j]);
    __syncthreads();
  }

  // ---- epilogue: bias (+residual) (+relu); C row = r+8*half, col = lane%16 ----
#pragma unroll
  for (int i = 0; i < 2; ++i)
#pragma unroll
    for (int j = 0; j < 2; ++j) {
      const int col = n0 + wn * 32 + j * 16 + mrow;
      const float bv = bias[col];
#pragma unroll
      for (int r = 0; r < 8; ++r) {
        const int row = m0 + wm * 32 + i * 16 + r + (half << 3);
        float x = acc[i][j][r] + bv;
        if (res)  x += res[(size_t)row * N + col];
        if (relu) x = fmaxf(x, 0.f);
        C[(size_t)row * N + col] = x;
      }
    }
}

// =====================================================================
// Flash attention (per b,h): O = softmax(Q K^T * scale) V
// Q,K,V,O: [B*rows, E], head h at cols h*64..h*64+63.
// WG = 256 thr = 8 waves; wave owns 16 query rows. S in 64-row blocks.
// Word-grid LDS (stride 36 words = 144B, 16B-aligned):
//   Qsw[qrow][dword]  (pairs along d)     -> A-frags, 2x b128
//   Ksw[srow][dword]  (pairs along d=k)   -> B-frags, 2x b128
//   Vsw[d][sword]     (pairs along s=k)   -> B-frags, 2x b128
//   Psw[wave][m][sword] (pairs along s=k) -> A-frags, 2x b128
// =====================================================================
#define FA_SB 64
#define WLD 36    // words per row (32 + 4 pad), 144B: 16B-aligned

__global__ __launch_bounds__(256)
void flash_attn_wmma(const float* __restrict__ Q, const float* __restrict__ Kg,
                     const float* __restrict__ Vg, float* __restrict__ O,
                     int Tq, int Skv, float scale)
{
  __shared__ unsigned int Qsw[128 * WLD];       // 18 KB
  __shared__ unsigned int Ksw[FA_SB * WLD];     //  9 KB
  __shared__ unsigned int Vsw[D_ * WLD];        //  9 KB
  __shared__ unsigned int Psw[8][16 * WLD];     // 18 KB

  const int tid  = threadIdx.x;
  const int wave = tid >> 5, lane = tid & 31;
  const int mrow = lane & 15, half = lane >> 4;
  const int b = blockIdx.y / H_, h = blockIdx.y % H_;
  const int t0 = blockIdx.x * 128;

  // ---- stage Q block (128 x 64), pairs along d ----
  {
    const int c = (tid & 15) * 4;
#pragma unroll
    for (int p = 0; p < 8; ++p) {
      const int r = p * 16 + (tid >> 4);
      const float4 f = *(const float4*)(Q + (size_t)(b * Tq + t0 + r) * E_ + h * D_ + c);
      uint2 w; w.x = pack_bf16(f.x, f.y); w.y = pack_bf16(f.z, f.w);
      *(uint2*)&Qsw[r * WLD + (c >> 1)] = w;
    }
  }
  __syncthreads();

  // ---- Q fragments (A-layout, 2 chunks over D=64), kept in regs ----
  v16bf qf[2];
  {
    const unsigned int* qb = &Qsw[(wave * 16 + mrow) * WLD];
#pragma unroll
    for (int ch = 0; ch < 2; ++ch)
      qf[ch] = load_frag4x2(qb + ch * 16 + (half << 2),
                            qb + ch * 16 + 8 + (half << 2));
  }

  float mr[8], lr[8];
  v8f oacc[4];
#pragma unroll
  for (int r = 0; r < 8; ++r) { mr[r] = -1e30f; lr[r] = 0.f; }
#pragma unroll
  for (int j = 0; j < 4; ++j)
#pragma unroll
    for (int r = 0; r < 8; ++r) oacc[j][r] = 0.f;

  for (int s0 = 0; s0 < Skv; s0 += FA_SB) {
    // ---- stage K block (64 x 64), pairs along d ----
    {
      const int c = (tid & 15) * 4;
#pragma unroll
      for (int p = 0; p < 4; ++p) {
        const int r = p * 16 + (tid >> 4);
        const float4 f = *(const float4*)(Kg + (size_t)(b * Skv + s0 + r) * E_ + h * D_ + c);
        uint2 w; w.x = pack_bf16(f.x, f.y); w.y = pack_bf16(f.z, f.w);
        *(uint2*)&Ksw[r * WLD + (c >> 1)] = w;
      }
    }
    // ---- stage V block transposed: Vsw[d][s-word], pairs along s ----
    {
      const int d  = tid & 63;
      const int wb = tid >> 6;                  // 0..3
#pragma unroll
      for (int p = 0; p < 8; ++p) {
        const int w = p * 4 + wb;               // s-word 0..31
        const size_t g = (size_t)(b * Skv + s0 + 2 * w) * E_ + h * D_ + d;
        const float f0 = Vg[g];
        const float f1 = Vg[g + E_];
        Vsw[d * WLD + w] = pack_bf16(f0, f1);
      }
    }
    __syncthreads();

    // ---- scores: 4 tiles of 16x16 over 64 source cols ----
    v8f sc[4];
#pragma unroll
    for (int nt = 0; nt < 4; ++nt) {
#pragma unroll
      for (int r = 0; r < 8; ++r) sc[nt][r] = 0.f;
      const unsigned int* kb = &Ksw[(nt * 16 + mrow) * WLD];  // B col n = src row
#pragma unroll
      for (int ch = 0; ch < 2; ++ch) {
        const v16bf kf = load_frag4x2(kb + ch * 16 + (half << 2), kb + ch * 16 + 8 + (half << 2));
        sc[nt] = wmma_bf16(qf[ch], kf, sc[nt]);
      }
#pragma unroll
      for (int r = 0; r < 8; ++r) sc[nt][r] *= scale;
    }

    // ---- online softmax (row r+8*half spread over 16-lane group) ----
    float mnew[8], corr[8], lad[8];
#pragma unroll
    for (int r = 0; r < 8; ++r) {
      float mx = fmaxf(fmaxf(sc[0][r], sc[1][r]), fmaxf(sc[2][r], sc[3][r]));
#pragma unroll
      for (int o = 8; o; o >>= 1) mx = fmaxf(mx, __shfl_xor(mx, o, 32));
      mnew[r] = fmaxf(mr[r], mx);
      corr[r] = __expf(mr[r] - mnew[r]);
      mr[r]   = mnew[r];
      lad[r]  = 0.f;
    }
#pragma unroll
    for (int nt = 0; nt < 4; ++nt)
#pragma unroll
      for (int r = 0; r < 8; ++r) {
        const float p = __expf(sc[nt][r] - mnew[r]);
        sc[nt][r] = p;
        lad[r] += p;
      }
#pragma unroll
    for (int r = 0; r < 8; ++r) {
      float s = lad[r];
#pragma unroll
      for (int o = 8; o; o >>= 1) s += __shfl_xor(s, o, 32);
      lr[r] = lr[r] * corr[r] + s;
    }
#pragma unroll
    for (int j = 0; j < 4; ++j)
#pragma unroll
      for (int r = 0; r < 8; ++r) oacc[j][r] *= corr[r];

    // ---- P: C-layout regs -> per-wave LDS (same-wave DS is in-order) ----
#pragma unroll
    for (int nt = 0; nt < 4; ++nt)
#pragma unroll
      for (int r = 0; r < 8; ++r) {
        __bf16* prow = (__bf16*)&Psw[wave][(r + (half << 3)) * WLD];
        prow[nt * 16 + mrow] = (__bf16)sc[nt][r];
      }

    // ---- O += P(16x64) @ V(64x64) ----
    const unsigned int* pb = &Psw[wave][mrow * WLD];
#pragma unroll
    for (int ch = 0; ch < 2; ++ch) {
      const v16bf pf = load_frag4x2(pb + ch * 16 + (half << 2),
                                    pb + ch * 16 + 8 + (half << 2));
#pragma unroll
      for (int j = 0; j < 4; ++j) {
        const v16bf vf = load_frag4x2(&Vsw[(j * 16 + mrow) * WLD + ch * 16 + (half << 2)], &Vsw[(j * 16 + mrow) * WLD + ch * 16 + 8 + (half << 2)]);
        oacc[j] = wmma_bf16(pf, vf, oacc[j]);
      }
    }
    __syncthreads();   // before next block restages Ksw/Vsw
  }

  // ---- normalize and write ----
#pragma unroll
  for (int j = 0; j < 4; ++j) {
    const int d = j * 16 + mrow;
#pragma unroll
    for (int r = 0; r < 8; ++r) {
      const int t = t0 + wave * 16 + r + (half << 3);
      O[(size_t)(b * Tq + t) * E_ + h * D_ + d] = oacc[j][r] / lr[r];
    }
  }
}

// =====================================================================
// LayerNorm over E=512, one row per 256-thread block (2 elems/thread).
// =====================================================================
__global__ __launch_bounds__(256)
void layernorm_512(const float* __restrict__ X, const float* __restrict__ g,
                   const float* __restrict__ be, float* __restrict__ Y)
{
  __shared__ float red[8];
  const int tid = threadIdx.x, lane = tid & 31, wave = tid >> 5;
  const float* x = X + (size_t)blockIdx.x * E_;
  const float v0 = x[tid], v1 = x[tid + 256];

  float s = v0 + v1;
#pragma unroll
  for (int o = 16; o; o >>= 1) s += __shfl_xor(s, o, 32);
  if (lane == 0) red[wave] = s;
  __syncthreads();
  float mean = 0.f;
#pragma unroll
  for (int i = 0; i < 8; ++i) mean += red[i];
  mean *= (1.f / E_);
  __syncthreads();

  const float d0 = v0 - mean, d1 = v1 - mean;
  float q = d0 * d0 + d1 * d1;
#pragma unroll
  for (int o = 16; o; o >>= 1) q += __shfl_xor(q, o, 32);
  if (lane == 0) red[wave] = q;
  __syncthreads();
  float var = 0.f;
#pragma unroll
  for (int i = 0; i < 8; ++i) var += red[i];
  var *= (1.f / E_);
  const float rstd = rsqrtf(var + EPS_);

  float* y = Y + (size_t)blockIdx.x * E_;
  y[tid]       = d0 * rstd * g[tid]       + be[tid];
  y[tid + 256] = d1 * rstd * g[tid + 256] + be[tid + 256];
}

// =====================================================================
// Host-side orchestration (graph-capture safe: launches on `stream` only)
// =====================================================================
extern "C" void kernel_launch(void* const* d_in, const int* in_sizes, int n_in,
                              void* d_out, int out_size, void* d_ws, size_t ws_size,
                              hipStream_t stream)
{
  (void)in_sizes; (void)n_in; (void)out_size; (void)ws_size;

  const float* tgt = (const float*)d_in[0];
  const float* mem = (const float*)d_in[1];
  const float* sWq = (const float*)d_in[2],  *sWk = (const float*)d_in[3];
  const float* sWv = (const float*)d_in[4],  *sWo = (const float*)d_in[5];
  const float* cWq = (const float*)d_in[6],  *cWk = (const float*)d_in[7];
  const float* cWv = (const float*)d_in[8],  *cWo = (const float*)d_in[9];
  const float* sbq = (const float*)d_in[10], *sbk = (const float*)d_in[11];
  const float* sbv = (const float*)d_in[12], *sbo = (const float*)d_in[13];
  const float* cbq = (const float*)d_in[14], *cbk = (const float*)d_in[15];
  const float* cbv = (const float*)d_in[16], *cbo = (const float*)d_in[17];
  const float* fW1 = (const float*)d_in[18], *fb1 = (const float*)d_in[19];
  const float* fW2 = (const float*)d_in[20], *fb2 = (const float*)d_in[21];
  const float* g1  = (const float*)d_in[22], *g2  = (const float*)d_in[23];
  const float* g3  = (const float*)d_in[24];
  const float* be1 = (const float*)d_in[25], *be2 = (const float*)d_in[26];
  const float* be3 = (const float*)d_in[27];
  float* out = (float*)d_out;

  const size_t SEG = (size_t)B_ * T_ * E_;   // 2,097,152 floats
  float* ws   = (float*)d_ws;
  float* q    = ws;
  float* kbuf = ws + 1 * SEG;
  float* vbuf = ws + 2 * SEG;
  float* attn = ws + 3 * SEG;
  float* tmp  = ws + 4 * SEG;
  float* x1   = ws + 5 * SEG;
  float* x2   = ws + 6 * SEG;
  float* ffh  = ws + 7 * SEG;                // needs 4*SEG (F=2048)

  const int Mr = B_ * T_;                    // 4096 rows (B*S identical)
  dim3 blk(256);
  dim3 gE(E_ / BN, Mr / BM);                 // GEMMs with N=512
  dim3 gF(F_ / BN, Mr / BM);                 // GEMM with N=2048
  dim3 gFA(T_ / 128, B_ * H_);               // flash attention

  // ---------------- self-attention block ----------------
  gemm_bf16_wmma<<<gE, blk, 0, stream>>>(tgt, sWq, sbq, nullptr, q,    Mr, E_, E_, 0);
  gemm_bf16_wmma<<<gE, blk, 0, stream>>>(tgt, sWk, sbk, nullptr, kbuf, Mr, E_, E_, 0);
  gemm_bf16_wmma<<<gE, blk, 0, stream>>>(tgt, sWv, sbv, nullptr, vbuf, Mr, E_, E_, 0);
  flash_attn_wmma<<<gFA, blk, 0, stream>>>(q, kbuf, vbuf, attn, T_, T_, SCALE_);
  gemm_bf16_wmma<<<gE, blk, 0, stream>>>(attn, sWo, sbo, tgt, tmp, Mr, E_, E_, 0);
  layernorm_512<<<Mr, blk, 0, stream>>>(tmp, g1, be1, x1);

  // ---------------- cross-attention block ----------------
  gemm_bf16_wmma<<<gE, blk, 0, stream>>>(x1,  cWq, cbq, nullptr, q,    Mr, E_, E_, 0);
  gemm_bf16_wmma<<<gE, blk, 0, stream>>>(mem, cWk, cbk, nullptr, kbuf, Mr, E_, E_, 0);
  gemm_bf16_wmma<<<gE, blk, 0, stream>>>(mem, cWv, cbv, nullptr, vbuf, Mr, E_, E_, 0);
  flash_attn_wmma<<<gFA, blk, 0, stream>>>(q, kbuf, vbuf, attn, T_, S_, SCALE_);
  gemm_bf16_wmma<<<gE, blk, 0, stream>>>(attn, cWo, cbo, x1, tmp, Mr, E_, E_, 0);
  layernorm_512<<<Mr, blk, 0, stream>>>(tmp, g2, be2, x2);

  // ---------------- FFN block ----------------
  gemm_bf16_wmma<<<gF, blk, 0, stream>>>(x2,  fW1, fb1, nullptr, ffh, Mr, F_, E_, 1);
  gemm_bf16_wmma<<<gE, blk, 0, stream>>>(ffh, fW2, fb2, x2,      tmp, Mr, E_, F_, 0);
  layernorm_512<<<Mr, blk, 0, stream>>>(tmp, g3, be3, out);
}
